// SGCLDGA_81423989997960
// MI455X (gfx1250) — matrix operand order repaired and checked by reading, not compiled, not run on hardware
//
#include <hip/hip_runtime.h>

#define D 128
#define NU 30000
#define NI 30000
#define BATCH 2048
#define TEMP_INV 5.0f          // 1/0.2
#define KEEPP 0.9f             // 1 - DROPOUT

typedef __attribute__((ext_vector_type(16))) __bf16 v16bf;
typedef __attribute__((ext_vector_type(8)))  float  v8f;

union FragBF { v16bf v; uint4 q[2]; };

// ---------------- WMMA fragment loaders (bf16, 16x16x32) -------------------
// A (16xK row-major): lane 0-15 -> M=lane, K={0..7,16..23}; lane 16-31 -> K={8..15,24..31}
__device__ __forceinline__ v16bf load_frag_a(const __bf16* A, int ld, int m0, int k0) {
  int lane = threadIdx.x & 31;
  int row  = m0 + (lane & 15);
  int kb   = (lane < 16) ? 0 : 8;
  const __bf16* p = A + (size_t)row * ld + k0 + kb;
  FragBF f;
  f.q[0] = *(const uint4*)(p);        // K = kb .. kb+7
  f.q[1] = *(const uint4*)(p + 16);   // K = kb+16 .. kb+23
  return f.v;
}
// B tile of X^T where X is [N,K] row-major: lane holds one column (= X row),
// K = {kb..kb+7, kb+16..kb+23} with kb in {0,8}.
__device__ __forceinline__ v16bf load_frag_b(const __bf16* X, int ld, int n0, int k0) {
  int lane = threadIdx.x & 31;
  int col  = n0 + (lane & 15);
  int kb   = (lane < 16) ? 0 : 8;
  const __bf16* p = X + (size_t)col * ld + k0 + kb;
  FragBF f;
  f.q[0] = *(const uint4*)(p);
  f.q[1] = *(const uint4*)(p + 16);
  return f.v;
}

__device__ __forceinline__ float warp_sum(float v) {
  #pragma unroll
  for (int m = 16; m > 0; m >>= 1) v += __shfl_xor(v, m, 32);
  return v;
}

// ---------------- utility kernels ------------------------------------------
__global__ void zerof_kernel(float* p, size_t n) {
  size_t t = (size_t)blockIdx.x * blockDim.x + threadIdx.x;
  size_t st = (size_t)gridDim.x * blockDim.x;
  for (; t < n; t += st) p[t] = 0.0f;
}

__global__ void pack_bf16_kernel(const float* __restrict__ s, __bf16* __restrict__ d, size_t n) {
  size_t t = (size_t)blockIdx.x * blockDim.x + threadIdx.x;
  size_t st = (size_t)gridDim.x * blockDim.x;
  for (; t < n; t += st) d[t] = (__bf16)s[t];
}

// WT[n*K + k] = W[k*N + n]   (pre-transpose weights so B-frag loads are contiguous)
__global__ void packT_kernel(const float* __restrict__ W, __bf16* __restrict__ WT, int K, int N) {
  int t = blockIdx.x * blockDim.x + threadIdx.x;
  if (t >= K * N) return;
  int n = t / K, k = t % K;
  WT[(size_t)n * K + k] = (__bf16)W[(size_t)k * N + n];
}

__global__ void gather_pack_kernel(const float* __restrict__ tab, const int* __restrict__ idx,
                                   __bf16* __restrict__ dst, int nrows, int dstride, int coff) {
  int t = blockIdx.x * blockDim.x + threadIdx.x;
  int r = t >> 5;
  if (r >= nrows) return;
  int lane = t & 31;
  float4 v = ((const float4*)(tab + (size_t)idx[r] * D))[lane];
  __bf16* d = dst + (size_t)r * dstride + coff + lane * 4;
  d[0] = (__bf16)v.x; d[1] = (__bf16)v.y; d[2] = (__bf16)v.z; d[3] = (__bf16)v.w;
}

__global__ void sum3_kernel(float* __restrict__ o, const float* __restrict__ a,
                            const float* __restrict__ b, const float* __restrict__ c, size_t n) {
  size_t t = (size_t)blockIdx.x * blockDim.x + threadIdx.x;
  size_t st = (size_t)gridDim.x * blockDim.x;
  for (; t < n; t += st) o[t] = a[t] + b[t] + c[t];
}

// ---------------- SpMM with deterministic edge dropout ---------------------
__device__ __forceinline__ unsigned hash_u32(unsigned x) {
  x ^= x >> 16; x *= 0x7feb352dU; x ^= x >> 15; x *= 0x846ca68bU; x ^= x >> 16; return x;
}

__global__ void spmm_kernel(const float* __restrict__ vals, const int* __restrict__ sidx,
                            const int* __restrict__ gidx, const float* __restrict__ X,
                            float* __restrict__ Y, int nE, unsigned seed) {
  int gid = blockIdx.x * blockDim.x + threadIdx.x;
  int lane = threadIdx.x & 31;
  int e = gid >> 5;
  if (e >= nE) return;
  unsigned h = hash_u32((unsigned)e * 0x9E3779B9u + seed * 0x85EBCA6Bu + 1u);
  const unsigned keep24 = (unsigned)(KEEPP * 16777216.0f);
  if ((h & 0xFFFFFFu) >= keep24) return;              // dropped edge (uniform per-wave)
  float v = vals[e] * (1.0f / KEEPP);
  int src = gidx[e], dst = sidx[e];
  __builtin_prefetch(X + (size_t)src * D, 0, 1);
  float4 x = ((const float4*)(X + (size_t)src * D))[lane];
  float* y = Y + (size_t)dst * D + lane * 4;
  atomicAdd(y + 0, v * x.x);
  atomicAdd(y + 1, v * x.y);
  atomicAdd(y + 2, v * x.z);
  atomicAdd(y + 3, v * x.w);
}

// ---------------- WMMA GEMM, A-fragment register reuse across NTILES -------
// B fragments for a subtile are preloaded as a group (one clause of b128 loads)
// so they overlap the previous subtile's WMMA chain.
template<int KD, int NTILES, bool RELU, bool OUTBF>
__global__ void wmma_gemm_kernel(const __bf16* __restrict__ A, const __bf16* __restrict__ BT,
                                 const float* __restrict__ bias, void* __restrict__ out, int ldo) {
  constexpr int KF = KD / 32;
  int tm = blockIdx.y;
  int tn0 = blockIdx.x * NTILES;
  int lane = threadIdx.x & 31;

  v16bf afr[KF];
  #pragma unroll
  for (int i = 0; i < KF; i++) afr[i] = load_frag_a(A, KD, tm * 16, i * 32);

  int colp = lane & 15;
  int rbase = tm * 16 + ((lane < 16) ? 0 : 8);

  #pragma unroll
  for (int t = 0; t < NTILES; t++) {
    v8f c = {};
    #pragma unroll
    for (int kc = 0; kc < KF; kc += 4) {       // chunked B preload (<=4 frags live)
      v16bf bfr[4];
      #pragma unroll
      for (int i = 0; i < 4; i++)
        if (kc + i < KF) bfr[i] = load_frag_b(BT, KD, (tn0 + t) * 16, (kc + i) * 32);
      #pragma unroll
      for (int i = 0; i < 4; i++)
        if (kc + i < KF)
          c = __builtin_amdgcn_wmma_f32_16x16x32_bf16(false, afr[kc + i], false, bfr[i],
                                                      (short)0, c, false, false);
    }
    int col = (tn0 + t) * 16 + colp;
    float bv = bias[col];
    #pragma unroll
    for (int r = 0; r < 8; r++) {
      float v = c[r] + bv;
      if (RELU) v = fmaxf(v, 0.0f);
      if (OUTBF) ((__bf16*)out)[(size_t)(rbase + r) * ldo + col] = (__bf16)v;
      else       ((float*)out)[(size_t)(rbase + r) * ldo + col] = v;
    }
  }
}

// ---------------- fused contrastive GEMM + exp + row-sum -------------------
// One wave: 16 batch rows x 64 table cols (4 subtiles, A frags held in regs).
// B fragments per subtile preloaded as one group; exp-sums accumulated in
// registers across subtiles; single cross-lane reduce + atomic pass.
__global__ void contrastive_kernel(const __bf16* __restrict__ A, const __bf16* __restrict__ X,
                                   float* __restrict__ rowsum) {
  int tm = blockIdx.y;
  int tn0 = blockIdx.x * 4;
  int lane = threadIdx.x & 31;

  v16bf afr[4];
  #pragma unroll
  for (int i = 0; i < 4; i++) afr[i] = load_frag_a(A, D, tm * 16, i * 32);

  float s[8];
  #pragma unroll
  for (int r = 0; r < 8; r++) s[r] = 0.0f;

  #pragma unroll
  for (int t = 0; t < 4; t++) {
    v16bf bfr[4];
    #pragma unroll
    for (int i = 0; i < 4; i++) bfr[i] = load_frag_b(X, D, (tn0 + t) * 16, i * 32);
    v8f c = {};
    #pragma unroll
    for (int i = 0; i < 4; i++)
      c = __builtin_amdgcn_wmma_f32_16x16x32_bf16(false, afr[i], false, bfr[i],
                                                  (short)0, c, false, false);
    #pragma unroll
    for (int r = 0; r < 8; r++) s[r] += __expf(c[r] * TEMP_INV);
  }

  #pragma unroll
  for (int m = 1; m < 16; m <<= 1) {          // reduce over 16 columns (stays in half-wave)
    #pragma unroll
    for (int r = 0; r < 8; r++) s[r] += __shfl_xor(s[r], m, 32);
  }
  if ((lane & 15) == 0) {
    int rbase = tm * 16 + ((lane < 16) ? 0 : 8);
    #pragma unroll
    for (int r = 0; r < 8; r++) atomicAdd(&rowsum[rbase + r], s[r]);
  }
}

// ---------------- small loss kernels ---------------------------------------
__global__ void dot_w3_kernel(const float* __restrict__ H, const float* __restrict__ w3,
                              const float* __restrict__ b3, float* __restrict__ out) {
  int r = blockIdx.x, lane = threadIdx.x;
  float4 h = ((const float4*)(H + (size_t)r * D))[lane];
  float4 w = ((const float4*)w3)[lane];
  float s = warp_sum(h.x * w.x + h.y * w.y + h.z * w.z + h.w * w.w);
  if (lane == 0) out[r] = s + b3[0];
}

__global__ void consistency_kernel(const float* __restrict__ Q, const float* __restrict__ K,
                                   float* __restrict__ acc) {
  int r = blockIdx.x, lane = threadIdx.x;
  float q = Q[(size_t)r * 32 + lane], k = K[(size_t)r * 32 + lane];
  float qq = warp_sum(q * q);
  float kk = warp_sum(k * k);
  float qk = warp_sum(q * k);
  if (lane == 0) {
    float dd = qk / (fmaxf(sqrtf(qq), 1e-12f) * fmaxf(sqrtf(kk), 1e-12f));
    float e = dd - 1.0f;
    atomicAdd(acc, e * e);
  }
}

__global__ void diag_dot_kernel(const float* __restrict__ Xa, const float* __restrict__ Xb,
                                const int* __restrict__ idx, float* __restrict__ acc) {
  int b = blockIdx.x, lane = threadIdx.x;
  int i = idx[b];
  float4 x = ((const float4*)(Xa + (size_t)i * D))[lane];
  float4 y = ((const float4*)(Xb + (size_t)i * D))[lane];
  float s = warp_sum(x.x * y.x + x.y * y.y + x.z * y.z + x.w * y.w);
  if (lane == 0) atomicAdd(acc, fminf(fmaxf(s * TEMP_INV, -5.0f), 5.0f));
}

__global__ void log_sum_kernel(const float* __restrict__ rs, float* __restrict__ acc, int n) {
  int t = blockIdx.x * blockDim.x + threadIdx.x;
  float v = (t < n) ? __logf(rs[t] + 1e-8f) : 0.0f;
  v = warp_sum(v);
  if ((threadIdx.x & 31) == 0) atomicAdd(acc, v);
}

__global__ void rank_loss_kernel(const float* __restrict__ ps, const float* __restrict__ ns,
                                 float* __restrict__ scal, int n) {
  int t = blockIdx.x * blockDim.x + threadIdx.x;
  float a = 0.f, b = 0.f, c = 0.f;
  if (t < n) {
    float p = ps[t], q = ns[t];
    a = log1pf(__expf(-p));       // softplus(-pos)
    b = log1pf(__expf(q));        // softplus(neg)
    c = log1pf(__expf(q - p));    // softplus(neg - pos)
  }
  a = warp_sum(a); b = warp_sum(b); c = warp_sum(c);
  if ((threadIdx.x & 31) == 0) {
    atomicAdd(&scal[1], a); atomicAdd(&scal[2], b); atomicAdd(&scal[3], c);
  }
}

__global__ void sumsq_kernel(const float* __restrict__ p, int n, float* __restrict__ acc) {
  int t = blockIdx.x * blockDim.x + threadIdx.x;
  int st = gridDim.x * blockDim.x;
  float s = 0.f;
  for (int i = t; i < n; i += st) { float v = p[i]; s += v * v; }
  s = warp_sum(s);
  if ((threadIdx.x & 31) == 0) atomicAdd(acc, s);
}

__global__ void finalize_kernel(const float* __restrict__ scal, float* __restrict__ out) {
  if (threadIdx.x != 0 || blockIdx.x != 0) return;
  float cons    = scal[0] / (float)NU;
  float loss_r  = (scal[1] + scal[2] + scal[3]) / (float)BATCH;
  float neg_s   = (scal[4] + scal[5]) / (float)BATCH;
  float pos_s   = (scal[6] + scal[7]) / (float)BATCH;
  float loss_sw = 0.2f * (neg_s - pos_s);          // LAMBDA_1 * loss_s
  float reg     = 1e-7f * scal[8];
  out[0] = reg + loss_sw + cons + loss_r;
  out[1] = loss_r;
  out[2] = loss_sw;
}

// ===========================================================================
extern "C" void kernel_launch(void* const* d_in, const int* in_sizes, int n_in,
                              void* d_out, int out_size, void* d_ws, size_t ws_size,
                              hipStream_t stream) {
  const float* EgG0 = (const float*)d_in[0];
  const float* EdG0 = (const float*)d_in[1];
  const float* EgS0 = (const float*)d_in[2];
  const float* EdS0 = (const float*)d_in[3];
  const float* qW1 = (const float*)d_in[4];  const float* qb1 = (const float*)d_in[5];
  const float* qW2 = (const float*)d_in[6];  const float* qb2 = (const float*)d_in[7];
  const float* kW1 = (const float*)d_in[8];  const float* kb1 = (const float*)d_in[9];
  const float* kW2 = (const float*)d_in[10]; const float* kb2 = (const float*)d_in[11];
  const float* cW1 = (const float*)d_in[12]; const float* cb1 = (const float*)d_in[13];
  const float* cW2 = (const float*)d_in[14]; const float* cb2 = (const float*)d_in[15];
  const float* cW3 = (const float*)d_in[16]; const float* cb3 = (const float*)d_in[17];
  const float* adj_vals = (const float*)d_in[18];
  const int* adj_rows = (const int*)d_in[19];
  const int* adj_cols = (const int*)d_in[20];
  const int* uids = (const int*)d_in[21];
  const int* iids = (const int*)d_in[22];
  const int* posI = (const int*)d_in[23];
  const int* negI = (const int*)d_in[24];
  const int E = in_sizes[18];

  const size_t TBL = (size_t)NU * D;

  // ---- workspace carve-out (256B aligned) ----
  size_t off = 0;
  auto af = [&](size_t n) -> float*  { float*  p = (float*) ((char*)d_ws + off); off = (off + n * 4 + 255) & ~(size_t)255; return p; };
  auto ah = [&](size_t n) -> __bf16* { __bf16* p = (__bf16*)((char*)d_ws + off); off = (off + n * 2 + 255) & ~(size_t)255; return p; };

  float* L1 = af(TBL * 4);                 // Eg1,Ed1,Gg1,Gd1 (contiguous w/ L2 for one zero-fill)
  float* L2 = af(TBL * 4);
  float* Eg = af(TBL); float* Ed = af(TBL); float* Gu = af(TBL); float* Gi = af(TBL);
  float* Qf = af((size_t)NU * 32); float* Kf = af((size_t)NU * 32);
  float* Hp2 = af((size_t)BATCH * D); float* Hn2 = af((size_t)BATCH * D);
  float* ps = af(BATCH); float* ns = af(BATCH);
  float* rsU = af(BATCH); float* rsI = af(BATCH); float* scal = af(64);  // rsU..scal contiguous
  __bf16* EgB = ah(TBL); __bf16* EdB = ah(TBL); __bf16* GuB = ah(TBL); __bf16* GiB = ah(TBL);
  __bf16* HqB = ah(TBL); __bf16* HkB = ah(TBL);
  __bf16* AuB = ah((size_t)BATCH * D); __bf16* AiB = ah((size_t)BATCH * D);
  __bf16* CpB = ah((size_t)BATCH * 2 * D); __bf16* CnB = ah((size_t)BATCH * 2 * D);
  __bf16* Hp1B = ah((size_t)BATCH * D); __bf16* Hn1B = ah((size_t)BATCH * D);
  __bf16* qW1T = ah(128 * 128); __bf16* qW2T = ah(32 * 128);
  __bf16* kW1T = ah(128 * 128); __bf16* kW2T = ah(32 * 128);
  __bf16* cW1T = ah(128 * 256); __bf16* cW2T = ah(128 * 128);

  // ---- zero accumulators (every call; ws is not re-poisoned between replays) ----
  zerof_kernel<<<4096, 256, 0, stream>>>(L1, TBL * 8);
  zerof_kernel<<<32, 256, 0, stream>>>(rsU, (size_t)(BATCH + BATCH + 64));

  // ---- 2-layer propagation: 8 SpMMs with dropout (seeds follow dkeys order) ----
  float *Eg1 = L1, *Ed1 = L1 + TBL, *Gg1 = L1 + 2 * TBL, *Gd1 = L1 + 3 * TBL;
  float *Eg2 = L2, *Ed2 = L2 + TBL, *Gg2 = L2 + 2 * TBL, *Gd2 = L2 + 3 * TBL;
  int spmmBlk = (int)(((size_t)E * 32 + 255) / 256);
  spmm_kernel<<<spmmBlk, 256, 0, stream>>>(adj_vals, adj_rows, adj_cols, EdG0, Eg1, E, 0);
  spmm_kernel<<<spmmBlk, 256, 0, stream>>>(adj_vals, adj_cols, adj_rows, EgG0, Ed1, E, 1);
  spmm_kernel<<<spmmBlk, 256, 0, stream>>>(adj_vals, adj_rows, adj_cols, EdS0, Gg1, E, 2);
  spmm_kernel<<<spmmBlk, 256, 0, stream>>>(adj_vals, adj_cols, adj_rows, EgS0, Gd1, E, 3);
  spmm_kernel<<<spmmBlk, 256, 0, stream>>>(adj_vals, adj_rows, adj_cols, Ed1, Eg2, E, 4);
  spmm_kernel<<<spmmBlk, 256, 0, stream>>>(adj_vals, adj_cols, adj_rows, Eg1, Ed2, E, 5);
  spmm_kernel<<<spmmBlk, 256, 0, stream>>>(adj_vals, adj_rows, adj_cols, Gd1, Gg2, E, 6);
  spmm_kernel<<<spmmBlk, 256, 0, stream>>>(adj_vals, adj_cols, adj_rows, Gg1, Gd2, E, 7);

  // ---- layer sums ----
  sum3_kernel<<<4096, 256, 0, stream>>>(Eg, EgG0, Eg1, Eg2, TBL);
  sum3_kernel<<<4096, 256, 0, stream>>>(Ed, EdG0, Ed1, Ed2, TBL);
  sum3_kernel<<<4096, 256, 0, stream>>>(Gu, EgS0, Gg1, Gg2, TBL);
  sum3_kernel<<<4096, 256, 0, stream>>>(Gi, EdS0, Gd1, Gd2, TBL);

  // ---- bf16 packs ----
  pack_bf16_kernel<<<4096, 256, 0, stream>>>(Eg, EgB, TBL);
  pack_bf16_kernel<<<4096, 256, 0, stream>>>(Ed, EdB, TBL);
  pack_bf16_kernel<<<4096, 256, 0, stream>>>(Gu, GuB, TBL);
  pack_bf16_kernel<<<4096, 256, 0, stream>>>(Gi, GiB, TBL);
  packT_kernel<<<64, 256, 0, stream>>>(qW1, qW1T, 128, 128);
  packT_kernel<<<16, 256, 0, stream>>>(qW2, qW2T, 128, 32);
  packT_kernel<<<64, 256, 0, stream>>>(kW1, kW1T, 128, 128);
  packT_kernel<<<16, 256, 0, stream>>>(kW2, kW2T, 128, 32);
  packT_kernel<<<128, 256, 0, stream>>>(cW1, cW1T, 256, 128);
  packT_kernel<<<64, 256, 0, stream>>>(cW2, cW2T, 128, 128);

  // ---- consistency branch: q/k MLPs (WMMA) + l2norm diag ----
  wmma_gemm_kernel<128, 4, true,  true ><<<dim3(2, NU / 16), 32, 0, stream>>>(GuB, qW1T, qb1, HqB, 128);
  wmma_gemm_kernel<128, 2, false, false><<<dim3(1, NU / 16), 32, 0, stream>>>(HqB, qW2T, qb2, Qf, 32);
  wmma_gemm_kernel<128, 4, true,  true ><<<dim3(2, NU / 16), 32, 0, stream>>>(EgB, kW1T, kb1, HkB, 128);
  wmma_gemm_kernel<128, 2, false, false><<<dim3(1, NU / 16), 32, 0, stream>>>(HkB, kW2T, kb2, Kf, 32);
  consistency_kernel<<<NU, 32, 0, stream>>>(Qf, Kf, &scal[0]);

  // ---- concat-MLP ranking losses (WMMA) ----
  gather_pack_kernel<<<(BATCH * 32 + 255) / 256, 256, 0, stream>>>(Eg, uids, CpB, BATCH, 256, 0);
  gather_pack_kernel<<<(BATCH * 32 + 255) / 256, 256, 0, stream>>>(Ed, posI, CpB, BATCH, 256, 128);
  gather_pack_kernel<<<(BATCH * 32 + 255) / 256, 256, 0, stream>>>(Eg, uids, CnB, BATCH, 256, 0);
  gather_pack_kernel<<<(BATCH * 32 + 255) / 256, 256, 0, stream>>>(Ed, negI, CnB, BATCH, 256, 128);
  wmma_gemm_kernel<256, 4, true, true ><<<dim3(2, BATCH / 16), 32, 0, stream>>>(CpB, cW1T, cb1, Hp1B, 128);
  wmma_gemm_kernel<128, 4, true, false><<<dim3(2, BATCH / 16), 32, 0, stream>>>(Hp1B, cW2T, cb2, Hp2, 128);
  dot_w3_kernel<<<BATCH, 32, 0, stream>>>(Hp2, cW3, cb3, ps);
  wmma_gemm_kernel<256, 4, true, true ><<<dim3(2, BATCH / 16), 32, 0, stream>>>(CnB, cW1T, cb1, Hn1B, 128);
  wmma_gemm_kernel<128, 4, true, false><<<dim3(2, BATCH / 16), 32, 0, stream>>>(Hn1B, cW2T, cb2, Hn2, 128);
  dot_w3_kernel<<<BATCH, 32, 0, stream>>>(Hn2, cW3, cb3, ns);
  rank_loss_kernel<<<BATCH / 256, 256, 0, stream>>>(ps, ns, scal, BATCH);

  // ---- InfoNCE: fused WMMA + exp + rowsum over [2048 x 30000] x 2 ----
  gather_pack_kernel<<<(BATCH * 32 + 255) / 256, 256, 0, stream>>>(Gu, uids, AuB, BATCH, 128, 0);
  gather_pack_kernel<<<(BATCH * 32 + 255) / 256, 256, 0, stream>>>(Gi, iids, AiB, BATCH, 128, 0);
  contrastive_kernel<<<dim3(NU / 64, BATCH / 16), 32, 0, stream>>>(AuB, EgB, rsU);
  contrastive_kernel<<<dim3(NI / 64, BATCH / 16), 32, 0, stream>>>(AiB, EdB, rsI);
  log_sum_kernel<<<BATCH / 256, 256, 0, stream>>>(rsU, &scal[4], BATCH);
  log_sum_kernel<<<BATCH / 256, 256, 0, stream>>>(rsI, &scal[5], BATCH);
  diag_dot_kernel<<<BATCH, 32, 0, stream>>>(Gu, Eg, uids, &scal[6]);
  diag_dot_kernel<<<BATCH, 32, 0, stream>>>(Gi, Ed, iids, &scal[7]);

  // ---- L2 regularization over all 18 params ----
  const float* prm[18] = {EgG0, EdG0, EgS0, EdS0, qW1, qb1, qW2, qb2,
                          kW1, kb1, kW2, kb2, cW1, cb1, cW2, cb2, cW3, cb3};
  const int prmN[18] = {(int)TBL, (int)TBL, (int)TBL, (int)TBL, 16384, 128, 4096, 32,
                        16384, 128, 4096, 32, 32768, 128, 16384, 128, 128, 1};
  for (int i = 0; i < 18; i++) {
    int n = prmN[i];
    int blk = (n + 255) / 256; if (blk > 1024) blk = 1024;
    sumsq_kernel<<<blk, 256, 0, stream>>>(prm[i], n, &scal[8]);
  }

  finalize_kernel<<<1, 1, 0, stream>>>(scal, (float*)d_out);
}
